// SpiderOpticNerve_64793876627802
// MI455X (gfx1250) — compile-verified, matrix operand order checked by reading, not run on hardware
//
#include <hip/hip_runtime.h>
#include <math.h>

typedef __attribute__((ext_vector_type(16))) _Float16 v16h;
typedef __attribute__((ext_vector_type(8)))  float    v8f;

#define K_IN 1536
#define NKS  (K_IN / 32)      // 48 K-steps of 32
#define M_TOT 8192            // 8*128*8 rows per stream
#define WAVES_PER_BLOCK 4
#define THREADS_PER_BLOCK (WAVES_PER_BLOCK * 32)
#define MT 2                  // 16-row M-tiles per wave (B fragment reuse)

// ---------------------------------------------------------------------------
// Pack W [K_IN, D] f32 (row-major) into per-lane B-fragment order, f16.
// Fragment layout for V_WMMA_F32_16X16X32_F16 B operand (mirrors 16-bit A
// layout with N in place of M):
//   lane l: column n = l%16, group g = l/16
//   halves j=0..7  -> K = k0 + g*8 + j
//   halves j=8..15 -> K = k0 + 16 + g*8 + (j-8)
// Packed linear index (halves): ((nt*NKS + ks)*32 + lane)*16 + j
// ---------------------------------------------------------------------------
template <int D>
__global__ void pack_w_kernel(const float* __restrict__ W, _Float16* __restrict__ P) {
    const int total = (D / 16) * NKS * 32 * 16;
    int t = blockIdx.x * blockDim.x + threadIdx.x;
    if (t >= total) return;
    const int j    = t & 15;
    const int lane = (t >> 4) & 31;
    const int ks   = (t >> 9) % NKS;
    const int nt   = t / (512 * NKS);
    const int g    = lane >> 4;
    const int n    = nt * 16 + (lane & 15);
    const int k0   = ks * 32;
    const int k    = (j < 8) ? (k0 + g * 8 + j) : (k0 + 16 + g * 8 + (j - 8));
    P[t] = (_Float16)W[(size_t)k * D + n];
}

// ---------------------------------------------------------------------------
// Fused GEMM (MT x 16-row tiles x D per wave, f16 WMMA, f32 accum) + bias +
// exact GELU + LayerNorm(D) + gamma/beta.
// Per K-step: batch all B-fragment loads (NT x 32B) and all A loads
// (MT x 4 x float4) up front so loadcnt waits count down instead of draining
// to zero before each WMMA; each B fragment feeds MT WMMAs.
// ---------------------------------------------------------------------------
template <int D>
__global__ __launch_bounds__(THREADS_PER_BLOCK)
void stream_kernel(const float* __restrict__ X, const _Float16* __restrict__ P,
                   const float* __restrict__ bias, const float* __restrict__ gamma,
                   const float* __restrict__ beta, float* __restrict__ Out) {
    constexpr int NT = D / 16;
    __shared__ float lds[WAVES_PER_BLOCK][MT * 16][D];

    const int lane = threadIdx.x & 31;
    const int wave = threadIdx.x >> 5;
    const int tile0 = (blockIdx.x * WAVES_PER_BLOCK + wave) * MT;
    const int m0   = tile0 * 16;
    const int g    = lane >> 4;      // lane group (K-octet select for A, row-half for C)
    const int r    = lane & 15;      // A-row within tile / C-column within n-tile

    const float* xrow[MT];
#pragma unroll
    for (int t = 0; t < MT; ++t)
        xrow[t] = X + (size_t)(m0 + t * 16 + r) * K_IN;

    v8f acc[MT][NT] = {};

    for (int ks = 0; ks < NKS; ++ks) {
        const int k0 = ks * 32;
        if (ks + 1 < NKS) {
#pragma unroll
            for (int t = 0; t < MT; ++t)
                __builtin_prefetch(xrow[t] + k0 + 32, 0, 0);  // global_prefetch
        }

        // ---- issue all B-fragment loads for this K-step first ----
        const v16h* bp = ((const v16h*)P) + ((size_t)ks * 32 + lane);
        v16h bf[NT];
#pragma unroll
        for (int nt = 0; nt < NT; ++nt)
            bf[nt] = bp[(size_t)nt * NKS * 32];

        // ---- issue all A raw loads (two 8-wide K runs per lane per tile) ----
        float4 f0[MT], f1[MT], f2[MT], f3[MT];
#pragma unroll
        for (int t = 0; t < MT; ++t) {
            f0[t] = *(const float4*)(xrow[t] + k0 + g * 8);
            f1[t] = *(const float4*)(xrow[t] + k0 + g * 8 + 4);
            f2[t] = *(const float4*)(xrow[t] + k0 + 16 + g * 8);
            f3[t] = *(const float4*)(xrow[t] + k0 + 16 + g * 8 + 4);
        }

        // ---- convert A to f16 fragments ----
        v16h a[MT];
#pragma unroll
        for (int t = 0; t < MT; ++t) {
            a[t][0]  = (_Float16)f0[t].x; a[t][1]  = (_Float16)f0[t].y;
            a[t][2]  = (_Float16)f0[t].z; a[t][3]  = (_Float16)f0[t].w;
            a[t][4]  = (_Float16)f1[t].x; a[t][5]  = (_Float16)f1[t].y;
            a[t][6]  = (_Float16)f1[t].z; a[t][7]  = (_Float16)f1[t].w;
            a[t][8]  = (_Float16)f2[t].x; a[t][9]  = (_Float16)f2[t].y;
            a[t][10] = (_Float16)f2[t].z; a[t][11] = (_Float16)f2[t].w;
            a[t][12] = (_Float16)f3[t].x; a[t][13] = (_Float16)f3[t].y;
            a[t][14] = (_Float16)f3[t].z; a[t][15] = (_Float16)f3[t].w;
        }

        // ---- WMMAs: each B fragment reused across MT tiles ----
#pragma unroll
        for (int nt = 0; nt < NT; ++nt) {
#pragma unroll
            for (int t = 0; t < MT; ++t) {
                acc[t][nt] = __builtin_amdgcn_wmma_f32_16x16x32_f16(
                    /*neg_a=*/false, a[t], /*neg_b=*/false, bf[nt],
                    /*c_mod=*/(short)0, acc[t][nt],
                    /*reuse_a=*/false, /*reuse_b=*/false);
            }
        }
    }

    // Epilogue part 1: bias + exact GELU, stage tiles to LDS.
    // C layout: lane = column r (within n-tile), element e -> row e + 8*g.
#pragma unroll
    for (int t = 0; t < MT; ++t) {
#pragma unroll
        for (int nt = 0; nt < NT; ++nt) {
            const int n  = nt * 16 + r;           // tile-local column
            const float bn = bias[n];
#pragma unroll
            for (int e = 0; e < 8; ++e) {
                float h  = acc[t][nt][e] + bn;
                float ge = 0.5f * h * (1.0f + erff(h * 0.70710678118654752f));
                lds[wave][t * 16 + e + 8 * g][n] = ge;
            }
        }
    }
    __syncthreads();

    // Epilogue part 2: LayerNorm over D. Two lanes per row (half row each),
    // cross-half combine via shfl_xor(16) within the wave32.
    const int row  = lane & 15;
    const int half = g;
#pragma unroll
    for (int t = 0; t < MT; ++t) {
        const float* hrow = &lds[wave][t * 16 + row][0];

        float s = 0.0f, sq = 0.0f;
#pragma unroll
        for (int i = 0; i < D / 2; ++i) {
            float v = hrow[half * (D / 2) + i];
            s  += v;
            sq += v * v;
        }
        s  += __shfl_xor(s, 16, 32);
        sq += __shfl_xor(sq, 16, 32);

        const float mu   = s * (1.0f / D);
        const float var  = sq * (1.0f / D) - mu * mu;
        const float rstd = rsqrtf(var + 1e-5f);

        float* orow = Out + (size_t)(m0 + t * 16 + row) * D;
#pragma unroll
        for (int i = 0; i < D / 2; ++i) {
            const int n = half * (D / 2) + i;
            const float v = (hrow[n] - mu) * rstd;
            orow[n] = v * gamma[n] + beta[n];
        }
    }
}

// ---------------------------------------------------------------------------
// Launch: 4 streams. Inputs per stream: x, w, b, g, beta (5 each).
// Outputs concatenated: principal(8192*64), motion(8192*32),
// texture(8192*32), nocturnal(8192*16).
// ---------------------------------------------------------------------------
extern "C" void kernel_launch(void* const* d_in, const int* in_sizes, int n_in,
                              void* d_out, int out_size, void* d_ws, size_t ws_size,
                              hipStream_t stream) {
    (void)in_sizes; (void)n_in; (void)out_size; (void)ws_size;

    const float* x0 = (const float*)d_in[0];
    const float* w0 = (const float*)d_in[1];
    const float* b0 = (const float*)d_in[2];
    const float* g0 = (const float*)d_in[3];
    const float* be0 = (const float*)d_in[4];

    const float* x1 = (const float*)d_in[5];
    const float* w1 = (const float*)d_in[6];
    const float* b1 = (const float*)d_in[7];
    const float* g1 = (const float*)d_in[8];
    const float* be1 = (const float*)d_in[9];

    const float* x2 = (const float*)d_in[10];
    const float* w2 = (const float*)d_in[11];
    const float* b2 = (const float*)d_in[12];
    const float* g2 = (const float*)d_in[13];
    const float* be2 = (const float*)d_in[14];

    const float* x3 = (const float*)d_in[15];
    const float* w3 = (const float*)d_in[16];
    const float* b3 = (const float*)d_in[17];
    const float* g3 = (const float*)d_in[18];
    const float* be3 = (const float*)d_in[19];

    // Packed f16 weight staging in workspace.
    const size_t h64 = (size_t)(64 / 16) * NKS * 32 * 16;  // 98304 halves
    const size_t h32 = (size_t)(32 / 16) * NKS * 32 * 16;  // 49152 halves
    const size_t h16 = (size_t)(16 / 16) * NKS * 32 * 16;  // 24576 halves
    _Float16* P0 = (_Float16*)d_ws;
    _Float16* P1 = P0 + h64;
    _Float16* P2 = P1 + h32;
    _Float16* P3 = P2 + h32;

    pack_w_kernel<64><<<(int)((h64 + 255) / 256), 256, 0, stream>>>(w0, P0);
    pack_w_kernel<32><<<(int)((h32 + 255) / 256), 256, 0, stream>>>(w1, P1);
    pack_w_kernel<32><<<(int)((h32 + 255) / 256), 256, 0, stream>>>(w2, P2);
    pack_w_kernel<16><<<(int)((h16 + 255) / 256), 256, 0, stream>>>(w3, P3);

    float* out = (float*)d_out;
    float* o0 = out;
    float* o1 = o0 + (size_t)M_TOT * 64;
    float* o2 = o1 + (size_t)M_TOT * 32;
    float* o3 = o2 + (size_t)M_TOT * 32;

    const int grid = (M_TOT / 16) / (WAVES_PER_BLOCK * MT);  // 64 blocks
    stream_kernel<64><<<grid, THREADS_PER_BLOCK, 0, stream>>>(x0, P0, b0, g0, be0, o0);
    stream_kernel<32><<<grid, THREADS_PER_BLOCK, 0, stream>>>(x1, P1, b1, g1, be1, o1);
    stream_kernel<32><<<grid, THREADS_PER_BLOCK, 0, stream>>>(x2, P2, b2, g2, be2, o2);
    stream_kernel<16><<<grid, THREADS_PER_BLOCK, 0, stream>>>(x3, P3, b3, g3, be3, o3);
}
